// MultiHeadAttention_34857954574735
// MI455X (gfx1250) — compile-verified, matrix operand order checked by reading, not run on hardware
//
#include <hip/hip_runtime.h>

// ---------------------------------------------------------------------------
// MHA forward for gfx1250 (MI455X).
// K0: f32 -> bf16 pre-convert of x and Wq/Wk/Wv (memory-bound).
// K1: QKV projection GEMM, pure bf16 WMMA, 32x64 tile per wave.
// K2: flash attention; K/V chunks staged to LDS via async global->LDS loads
//     shared by the 4 waves of a block; online softmax; bf16 WMMA.
// ---------------------------------------------------------------------------

typedef __attribute__((ext_vector_type(16))) __bf16 v16bf;
typedef __attribute__((ext_vector_type(2)))  __bf16 v2bf;
typedef __attribute__((ext_vector_type(8)))  float  v8f;
typedef __attribute__((ext_vector_type(4)))  int    v4i;

#define BATCH   2
#define S_LEN   2048
#define DMODEL  1024
#define NHEAD   16
#define HDIM    64
#define SCALE   0.03125f      /* 1/sqrt(1024) */
#define NEGV    (-1e9f)

// ---------------- f32 -> bf16 helpers -------------------------------------
static __device__ __forceinline__ unsigned short f2bf_bits(float f) {
  union { float f; unsigned u; } c; c.f = f;
  unsigned u = c.u;
  unsigned r = (u + 0x7FFFu + ((u >> 16) & 1u)) >> 16;   // RNE
  return (unsigned short)r;
}

#if defined(__has_builtin)
#if __has_builtin(__builtin_amdgcn_cvt_pk_bf16_f32)
#define HAVE_PK_BF16 1
#endif
#if __has_builtin(__builtin_amdgcn_global_load_async_to_lds_b128)
#define HAVE_ASYNC_LDS 1
#endif
#if __has_builtin(__builtin_amdgcn_s_wait_asynccnt)
#define HAVE_WAIT_ASYNC 1
#endif
#endif

static __device__ __forceinline__ unsigned pack_bf16(float a, float b) {
#ifdef HAVE_PK_BF16
  union { v2bf v; unsigned u; } c;
  c.v = __builtin_amdgcn_cvt_pk_bf16_f32(a, b);
  return c.u;
#else
  return (unsigned)f2bf_bits(a) | ((unsigned)f2bf_bits(b) << 16);
#endif
}

// ---------------- fragment helpers -----------------------------------------
union FragB16 { unsigned u[8]; v16bf v; };   // 8 VGPRs = 16 bf16 (A or B frag)
union AccF    { v8f v; float f[8]; };        // 8 VGPRs = 16x16 f32 C/D frag

// K-pair base for VGPR v (0..7) in the 16-bit A/B fragment layout:
// VGPR v holds K = kbase, kbase+1 packed low/high.
static __device__ __forceinline__ int kbase(int v, int half) {
  return 2 * v + ((v & 4) ? 8 : 0) + 8 * half;
}

static __device__ __forceinline__ v8f zero_v8f() {
  v8f z;
#pragma unroll
  for (int i = 0; i < 8; ++i) z[i] = 0.0f;
  return z;
}

static __device__ __forceinline__ v8f wmma_bf16(const FragB16& a, const FragB16& b, v8f c) {
  return __builtin_amdgcn_wmma_f32_16x16x32_bf16(false, a.v, false, b.v,
                                                 (short)0, c, false, false);
}

// ---------------- async global->LDS staging --------------------------------
static __device__ __forceinline__ void stage_b128(const unsigned short* gsrc,
                                                  unsigned short* ldst) {
#ifdef HAVE_ASYNC_LDS
  __builtin_amdgcn_global_load_async_to_lds_b128(
      (__attribute__((address_space(1))) v4i*)gsrc,
      (__attribute__((address_space(3))) v4i*)ldst, 0, 0);
#else
  *(uint4*)ldst = *(const uint4*)gsrc;
#endif
}

static __device__ __forceinline__ void wait_async() {
#ifdef HAVE_ASYNC_LDS
#ifdef HAVE_WAIT_ASYNC
  __builtin_amdgcn_s_wait_asynccnt(0);
#else
  asm volatile("s_wait_asynccnt 0x0" ::: "memory");
#endif
#endif
}

// ---------------------------------------------------------------------------
// Kernel 0: elementwise f32 -> bf16 (4 floats / thread).
// ---------------------------------------------------------------------------
__global__ __launch_bounds__(256)
void cvt_bf16_kernel(const float* __restrict__ src,
                     unsigned short* __restrict__ dst, int n4) {
  const int i = blockIdx.x * blockDim.x + threadIdx.x;
  if (i < n4) {
    const float4 f = ((const float4*)src)[i];
    ((uint2*)dst)[i] = make_uint2(pack_bf16(f.x, f.y), pack_bf16(f.z, f.w));
  }
}

// ---------------------------------------------------------------------------
// Kernel 1: Q/K/V projection.  grid = (32, 16, 3), block = 128 (4 waves).
// Wave tile: 32 rows x 64 cols (one head's HD span). Pure bf16 WMMA loop.
// ---------------------------------------------------------------------------
__global__ __launch_bounds__(128)
void qkv_proj_kernel(const unsigned short* __restrict__ Xb,
                     const unsigned short* __restrict__ Wqb,
                     const unsigned short* __restrict__ Wkb,
                     const unsigned short* __restrict__ Wvb,
                     const float* __restrict__ bq,
                     const float* __restrict__ bk,
                     const float* __restrict__ bv,
                     unsigned short* __restrict__ Qh,
                     unsigned short* __restrict__ Kh,
                     unsigned short* __restrict__ Vt) {
  const int lane = threadIdx.x & 31;
  const int wave = threadIdx.x >> 5;
  const int half = lane >> 4;
  const int l16  = lane & 15;

  const int mtile = blockIdx.x * 4 + wave;     // 32-row tile of B*S rows
  const int hidx  = blockIdx.y;                // head index (64 output cols)
  const int z     = blockIdx.z;                // 0:Q 1:K 2:V

  const unsigned short* W = (z == 0) ? Wqb : (z == 1) ? Wkb : Wvb;
  const float* bias       = (z == 0) ? bq  : (z == 1) ? bk  : bv;

  const int n0 = hidx * HDIM;

  AccF acc[2][4];
#pragma unroll
  for (int mt = 0; mt < 2; ++mt)
#pragma unroll
    for (int t = 0; t < 4; ++t) acc[mt][t].v = zero_v8f();

  for (int k0 = 0; k0 < DMODEL; k0 += 32) {
    FragB16 a[2];
#pragma unroll
    for (int mt = 0; mt < 2; ++mt) {
      const int m_row = mtile * 32 + mt * 16 + l16;
#pragma unroll
      for (int v = 0; v < 8; ++v)
        a[mt].u[v] = *(const unsigned*)(Xb + (size_t)m_row * DMODEL +
                                        k0 + kbase(v, half));
    }
#pragma unroll
    for (int t = 0; t < 4; ++t) {
      FragB16 bf;               // B[k][n] = W[n][k], k contiguous
      const int n = n0 + t * 16 + l16;
#pragma unroll
      for (int v = 0; v < 8; ++v)
        bf.u[v] = *(const unsigned*)(W + (size_t)n * DMODEL +
                                     k0 + kbase(v, half));
#pragma unroll
      for (int mt = 0; mt < 2; ++mt)
        acc[mt][t].v = wmma_bf16(a[mt], bf, acc[mt][t].v);
    }
  }

  // Store: Q/K as [B,H,S,HD] bf16; V transposed as [B,H,HD,S] bf16.
#pragma unroll
  for (int mt = 0; mt < 2; ++mt) {
#pragma unroll
    for (int t = 0; t < 4; ++t) {
      const int   n    = n0 + t * 16 + l16;
      const float bval = bias[n];
      const int   hd   = t * 16 + l16;
#pragma unroll
      for (int r = 0; r < 8; ++r) {
        const int m_glob = mtile * 32 + mt * 16 + r + 8 * half;
        const int b = m_glob / S_LEN;
        const int s = m_glob % S_LEN;
        const unsigned short val = f2bf_bits(acc[mt][t].f[r] + bval);
        const size_t bh = (size_t)b * NHEAD + hidx;
        if (z == 0)      Qh[(bh * S_LEN + s) * HDIM + hd] = val;
        else if (z == 1) Kh[(bh * S_LEN + s) * HDIM + hd] = val;
        else             Vt[(bh * HDIM + hd) * S_LEN + s] = val;
      }
    }
  }
}

// ---------------------------------------------------------------------------
// Kernel 2: flash attention.  grid = (1024), block = 128 (4 waves).
// The 4 waves of a block share one (b,h) and walk the same key chunks, so
// each 32-key K chunk (4KB) and Vt chunk (4KB) is staged into LDS once per
// block via async global->LDS b128 loads.
// ---------------------------------------------------------------------------
__global__ __launch_bounds__(128)
void attn_kernel(const unsigned short* __restrict__ Qh,
                 const unsigned short* __restrict__ Kh,
                 const unsigned short* __restrict__ Vt,
                 const unsigned char* __restrict__ mask,
                 float* __restrict__ Out) {
  __shared__ unsigned short Klds[32 * 64];      // [key_in_chunk][hd]
  __shared__ unsigned short Vlds[64 * 32];      // [hd][key_in_chunk]
  __shared__ unsigned short Plds[4][16 * 32];   // per-wave P tile

  const int tid  = threadIdx.x;
  const int lane = tid & 31;
  const int wave = tid >> 5;
  const int half = lane >> 4;
  const int l16  = lane & 15;

  const int qt = blockIdx.x * 4 + wave;         // global q-tile id
  const int bh = qt >> 7;                       // 128 q-tiles per (b,h)
  const int q0 = (qt & 127) * 16;
  const int b  = bh >> 4;
  const int h  = bh & 15;

  const unsigned short* Qbh = Qh + (size_t)bh * S_LEN * HDIM;
  const unsigned short* Kbh = Kh + (size_t)bh * S_LEN * HDIM;
  const unsigned short* Vbh = Vt + (size_t)bh * HDIM * S_LEN;
  const unsigned char*  mrow = mask + ((size_t)b * S_LEN + q0) * S_LEN;

  // Q fragments: 16x64 as two k=32 A-fragments, resident in VGPRs.
  FragB16 qf[2];
#pragma unroll
  for (int g = 0; g < 2; ++g)
#pragma unroll
    for (int v = 0; v < 8; ++v)
      qf[g].u[v] = *(const unsigned*)(Qbh + (size_t)(q0 + l16) * HDIM +
                                      g * 32 + kbase(v, half));

  float mrun[8], lrun[8];
  AccF  oacc[4];
#pragma unroll
  for (int r = 0; r < 8; ++r) { mrun[r] = -3.0e38f; lrun[r] = 0.0f; }
#pragma unroll
  for (int t = 0; t < 4; ++t) oacc[t].v = zero_v8f();

  for (int c = 0; c < S_LEN; c += 32) {
    __syncthreads();   // previous chunk fully consumed before overwrite

    // ---- cooperative staging: 256 x 16B segments each for K and Vt ----
#pragma unroll
    for (int i = 0; i < 2; ++i) {
      const int s = tid + i * 128;                        // 0..255
      stage_b128(Kbh + (size_t)(c + (s >> 3)) * HDIM + (s & 7) * 8,
                 &Klds[s * 8]);
      stage_b128(Vbh + (size_t)(s >> 2) * S_LEN + c + (s & 3) * 8,
                 &Vlds[s * 8]);
    }
    wait_async();
    __syncthreads();

    // ---- scores: two 16x16 tiles (keys c..c+15, c+16..c+31) ----
    AccF sacc[2];
#pragma unroll
    for (int st = 0; st < 2; ++st) {
      FragB16 kf0, kf1;    // B[k=hd][n=key] = K[key][hd] from LDS
#pragma unroll
      for (int v = 0; v < 8; ++v) {
        const unsigned short* p = &Klds[(st * 16 + l16) * HDIM + kbase(v, half)];
        kf0.u[v] = *(const unsigned*)p;
        kf1.u[v] = *(const unsigned*)(p + 32);
      }
      v8f sv = zero_v8f();
      sv = wmma_bf16(qf[0], kf0, sv);
      sv = wmma_bf16(qf[1], kf1, sv);
      sacc[st].v = sv;
      const int key0 = c + st * 16;
#pragma unroll
      for (int r = 0; r < 8; ++r) {
        const int mq = r + 8 * half;
        const unsigned char mk = mrow[(size_t)mq * S_LEN + key0 + l16];
        sacc[st].f[r] = mk ? NEGV : sacc[st].f[r] * SCALE;
      }
    }

    // ---- online softmax (row stats across each 16-lane group) ----
    float corr[8];
#pragma unroll
    for (int r = 0; r < 8; ++r) {
      float mr = fmaxf(sacc[0].f[r], sacc[1].f[r]);
#pragma unroll
      for (int off = 1; off < 16; off <<= 1)
        mr = fmaxf(mr, __shfl_xor(mr, off, 32));
      const float mn = fmaxf(mrun[r], mr);
      corr[r] = __expf(mrun[r] - mn);
      mrun[r] = mn;
      const float p0 = __expf(sacc[0].f[r] - mn);
      const float p1 = __expf(sacc[1].f[r] - mn);
      sacc[0].f[r] = p0; sacc[1].f[r] = p1;
      float rs = p0 + p1;
#pragma unroll
      for (int off = 1; off < 16; off <<= 1)
        rs += __shfl_xor(rs, off, 32);
      lrun[r] = lrun[r] * corr[r] + rs;
    }

    // ---- P: D layout -> A layout via wave-private LDS tile ----
    unsigned short* P = Plds[wave];
#pragma unroll
    for (int r = 0; r < 8; ++r) {
      const int mq = r + 8 * half;
      P[mq * 32 + l16]      = f2bf_bits(sacc[0].f[r]);
      P[mq * 32 + 16 + l16] = f2bf_bits(sacc[1].f[r]);
    }
    FragB16 pf;
#pragma unroll
    for (int v = 0; v < 8; ++v)
      pf.u[v] = *(const unsigned*)(P + l16 * 32 + kbase(v, half));

    // ---- O += P @ V (Vt in LDS: row=hd, col=key-in-chunk) ----
#pragma unroll
    for (int t = 0; t < 4; ++t) {
#pragma unroll
      for (int r = 0; r < 8; ++r) oacc[t].f[r] *= corr[r];
      FragB16 vf;
#pragma unroll
      for (int v = 0; v < 8; ++v)
        vf.u[v] = *(const unsigned*)&Vlds[(t * 16 + l16) * 32 + kbase(v, half)];
      oacc[t].v = wmma_bf16(pf, vf, oacc[t].v);
    }
  }

  // ---- finalize + store: out[b][s][h*HD + hd], f32 ----
#pragma unroll
  for (int t = 0; t < 4; ++t)
#pragma unroll
    for (int r = 0; r < 8; ++r) {
      const int s = q0 + r + 8 * half;
      Out[((size_t)b * S_LEN + s) * DMODEL + h * HDIM + t * 16 + l16] =
          oacc[t].f[r] / lrun[r];
    }
}

// ---------------------------------------------------------------------------
extern "C" void kernel_launch(void* const* d_in, const int* in_sizes, int n_in,
                              void* d_out, int out_size, void* d_ws, size_t ws_size,
                              hipStream_t stream) {
  (void)in_sizes; (void)n_in; (void)out_size; (void)ws_size;
  const float*         x    = (const float*)d_in[0];
  const unsigned char* mask = (const unsigned char*)d_in[1];
  const float* Wq = (const float*)d_in[2];
  const float* bq = (const float*)d_in[3];
  const float* Wk = (const float*)d_in[4];
  const float* bk = (const float*)d_in[5];
  const float* Wv = (const float*)d_in[6];
  const float* bv = (const float*)d_in[7];
  float* out = (float*)d_out;

  const size_t eQKV = (size_t)BATCH * NHEAD * S_LEN * HDIM;  // 4M
  const size_t eX   = (size_t)BATCH * S_LEN * DMODEL;        // 4M
  const size_t eW   = (size_t)DMODEL * DMODEL;               // 1M
  unsigned short* Qh  = (unsigned short*)d_ws;
  unsigned short* Kh  = Qh  + eQKV;
  unsigned short* Vt  = Kh  + eQKV;
  unsigned short* Xb  = Vt  + eQKV;
  unsigned short* Wqb = Xb  + eX;
  unsigned short* Wkb = Wqb + eW;
  unsigned short* Wvb = Wkb + eW;

  // K0: pre-convert to bf16 (memory-bound).
  cvt_bf16_kernel<<<(int)(eX / 4 + 255) / 256, 256, 0, stream>>>(x, Xb, (int)(eX / 4));
  cvt_bf16_kernel<<<(int)(eW / 4 + 255) / 256, 256, 0, stream>>>(Wq, Wqb, (int)(eW / 4));
  cvt_bf16_kernel<<<(int)(eW / 4 + 255) / 256, 256, 0, stream>>>(Wk, Wkb, (int)(eW / 4));
  cvt_bf16_kernel<<<(int)(eW / 4 + 255) / 256, 256, 0, stream>>>(Wv, Wvb, (int)(eW / 4));

  // K1: (BS/32)/4 = 32 blocks, 16 heads, 3 matrices.
  qkv_proj_kernel<<<dim3(32, 16, 3), 128, 0, stream>>>(
      Xb, Wqb, Wkb, Wvb, bq, bk, bv, Qh, Kh, Vt);

  // K2: B*H*(S/16)/4 = 1024 blocks, 4 waves each.
  attn_kernel<<<dim3(1024), 128, 0, stream>>>(Qh, Kh, Vt, mask, out);
}
